// Attention_39719857553906
// MI455X (gfx1250) — matrix operand order repaired and checked by reading, not compiled, not run on hardware
//
#include <hip/hip_runtime.h>

#define BATCH 4
#define DIM   192
#define QKV   (3*DIM)      // 576
#define HEADS 8
#define CH    (DIM/HEADS)  // 24
#define IMG   128
#define HW    (IMG*IMG)    // 16384

typedef __attribute__((ext_vector_type(16))) __bf16 v16bf;
typedef __attribute__((ext_vector_type(8)))  float  v8f;
typedef __attribute__((ext_vector_type(4)))  unsigned int v4u;
typedef __attribute__((ext_vector_type(8)))  int    v8i;
typedef __attribute__((ext_vector_type(4)))  int    v4i;

__device__ __forceinline__ v8f wmma_bf16(v16bf a, v16bf b, v8f c) {
    // D = A(16x32) * B(32x16) + C, fp32 accumulate
    return __builtin_amdgcn_wmma_f32_16x16x32_bf16(
        /*neg_a=*/false, a, /*neg_b=*/false, b,
        /*c_mod=*/(short)0, c, /*reuse_a=*/false, /*reuse_b=*/false);
}

// ---- fragment loaders (wave32), contiguous k order k = 16*(lane/16) + s in A and B ----
// A: 16(M) x 32(K). Lane L<16: M=L, K 0..15; L>=16: M=L-16, K 16..31.
__device__ __forceinline__ v16bf load_a_f32(const float* A, size_t lda, int lane) {
    v16bf r; int m = lane & 15, kb = (lane >> 4) * 16;
    const float* row = A + (size_t)m * lda + kb;
#pragma unroll
    for (int i = 0; i < 8; i++) { r[i] = (__bf16)row[i]; r[8 + i] = (__bf16)row[8 + i]; }
    return r;
}
__device__ __forceinline__ v16bf load_a_bf16(const __bf16* A, size_t lda, int lane) {
    v16bf r; int m = lane & 15, kb = (lane >> 4) * 16;
    const __bf16* row = A + (size_t)m * lda + kb;
#pragma unroll
    for (int i = 0; i < 8; i++) { r[i] = row[i]; r[8 + i] = row[8 + i]; }
    return r;
}
__device__ __forceinline__ v16bf load_a_bf16_m(const __bf16* A, size_t lda, int lane, int mrows) {
    v16bf r; int m = lane & 15, kb = (lane >> 4) * 16;
    if (m < mrows) {
        const __bf16* row = A + (size_t)m * lda + kb;
#pragma unroll
        for (int i = 0; i < 8; i++) { r[i] = row[i]; r[8 + i] = row[8 + i]; }
    } else {
#pragma unroll
        for (int i = 0; i < 16; i++) r[i] = (__bf16)0.0f;
    }
    return r;
}
// B: 32(K) x 16(N): lane L: N=L&15; L<16 holds K=0..15, L>=16 holds K=16..31.
// LDS-staged B panel (row-major [K][N]) in fp32 / bf16:
__device__ __forceinline__ v16bf load_b_lds_f32(const float* Bs, int ldb, int lane) {
    v16bf r; int n = lane & 15, kb = (lane >> 4) * 16;
    const float* p = Bs + kb * ldb + n;
#pragma unroll
    for (int i = 0; i < 16; i++) r[i] = (__bf16)p[i * ldb];
    return r;
}
__device__ __forceinline__ v16bf load_b_lds_bf16(const __bf16* Bs, int ldb, int lane) {
    v16bf r; int n = lane & 15, kb = (lane >> 4) * 16;
    const __bf16* p = Bs + kb * ldb + n;
#pragma unroll
    for (int i = 0; i < 16; i++) r[i] = p[i * ldb];
    return r;
}
// B from [N][K] row-major source (K^T access), contiguous per lane, with column masking.
__device__ __forceinline__ v16bf load_bT_bf16_m(const __bf16* Bs, size_t ldn, int lane, int ncols) {
    v16bf r; int n = lane & 15, kb = (lane >> 4) * 16;
    if (n < ncols) {
        const __bf16* p = Bs + (size_t)n * ldn + kb;
#pragma unroll
        for (int i = 0; i < 16; i++) r[i] = p[i];
    } else {
#pragma unroll
        for (int i = 0; i < 16; i++) r[i] = (__bf16)0.0f;
    }
    return r;
}

// Flat shared-memory pointer -> LDS byte address (ISA 10.2: LDS flat addr truncates to addr[31:0]).
__device__ __forceinline__ unsigned lds_off_u32(const void* p) {
    return (unsigned)(unsigned long long)p;
}

// ---- Tensor Data Mover: 2-D tile load Global -> LDS (D# per CDNA5 ISA) ----
// data_size_code: 0=1B,1=2B,2=4B,3=8B. Tile: tile_d0 (contig) x tile_d1 rows, row stride stride0.
__device__ __forceinline__ void tdm_load_2d(unsigned lds_addr, const void* gaddr,
                                            unsigned tensor_d0, unsigned tensor_d1,
                                            unsigned tile_d0, unsigned tile_d1,
                                            unsigned long long stride0, int data_size_code) {
    unsigned long long ga = (unsigned long long)gaddr;
    v4u g0;
    g0[0] = 1u;                                                   // count=1, user-mode D#
    g0[1] = lds_addr;                                             // lds_addr[31:0]
    g0[2] = (unsigned)ga;                                         // global_addr[31:0]
    g0[3] = (unsigned)((ga >> 32) & 0x01FFFFFFu) | (2u << 30);    // global_addr[56:32] | type=2
    v8i g1;
    g1[0] = (int)((unsigned)data_size_code << 16);                // wg_mask=0, data_size
    g1[1] = (int)((tensor_d0 & 0xFFFFu) << 16);                   // atomic_barrier=0 | dim0 lo
    g1[2] = (int)((tensor_d0 >> 16) | ((tensor_d1 & 0xFFFFu) << 16));
    g1[3] = (int)((tensor_d1 >> 16) | (tile_d0 << 16));
    g1[4] = (int)(tile_d1 & 0xFFFFu);                             // tile_dim1 | tile_dim2=0
    g1[5] = (int)(unsigned)(stride0 & 0xFFFFFFFFu);               // dim0 stride lo
    g1[6] = (int)(unsigned)((stride0 >> 32) & 0xFFFFu);           // dim0 stride hi | dim1 stride=0
    g1[7] = 0;
    v4i zz = {0, 0, 0, 0};
#if __clang_major__ >= 23
    v8i z8 = {0, 0, 0, 0, 0, 0, 0, 0};
    __builtin_amdgcn_tensor_load_to_lds(g0, g1, zz, zz, z8, 0);
#else
    __builtin_amdgcn_tensor_load_to_lds(g0, g1, zz, zz, 0);
#endif
}

// =============== Kernel 1: fused QKV GEMM (x -> K,V rows; y -> Q rows), bf16 out ===============
// grid: (HW/64, 9, BATCH), block 128 (4 waves). Block tile 64(M) x 64(N); B panel (192x64 f32,
// 48 KB) staged once per block by the TDM and shared by all 4 waves.
__global__ void qkv_gemm_kernel(const float* __restrict__ x, const float* __restrict__ y,
                                const float* __restrict__ qkv_w, __bf16* __restrict__ pre) {
    __shared__ float Bpanel[DIM * 64];          // 48 KB
    int lane = threadIdx.x & 31, wave = threadIdx.x >> 5;
    int mtile = blockIdx.y * 4 + wave;          // 0..35
    int o0 = mtile * 16;
    int n0 = blockIdx.x * 64;
    int b  = blockIdx.z;
    const float* src = (blockIdx.y < 3) ? y : x;    // mtiles 0..11 = Q (from y); 12..35 = K,V (from x)
    const float* gsrc = src + (size_t)b * DIM * HW + n0;
    if (threadIdx.x == 0) {
        tdm_load_2d(lds_off_u32(Bpanel), gsrc, (unsigned)HW, (unsigned)DIM,
                    64u, (unsigned)DIM, (unsigned long long)HW * sizeof(float) / sizeof(float) == 0 ? 0ull : (unsigned long long)HW, 2);
        __builtin_amdgcn_s_wait_tensorcnt((short)0);
    }
    __builtin_prefetch(qkv_w + (size_t)o0 * DIM, 0, 3);
    __syncthreads();
    v8f z = {0.f,0.f,0.f,0.f,0.f,0.f,0.f,0.f};
    v8f acc[4] = {z, z, z, z};
    for (int k0 = 0; k0 < DIM; k0 += 32) {
        v16bf a = load_a_f32(qkv_w + (size_t)o0 * DIM + k0, DIM, lane);
#pragma unroll
        for (int t = 0; t < 4; t++) {
            v16bf bb = load_b_lds_f32(&Bpanel[k0 * 64 + t * 16], 64, lane);
            acc[t] = wmma_bf16(a, bb, acc[t]);
        }
    }
    int n = lane & 15, mb = (lane >> 4) * 8;
    __bf16* outp = pre + ((size_t)b * QKV + o0) * HW + n0;
#pragma unroll
    for (int t = 0; t < 4; t++)
#pragma unroll
        for (int j = 0; j < 8; j++)
            outp[(size_t)(mb + j) * HW + t * 16 + n] = (__bf16)acc[t][j];
}

// =============== Kernel 2: depthwise 3x3 (SAME, zero pad) + sum-of-squares for q/k ===============
__global__ void dwconv_kernel(const __bf16* __restrict__ pre, const float* __restrict__ dw_w,
                              __bf16* __restrict__ out, float* __restrict__ norms) {
    __shared__ float red[4];
    int col = threadIdx.x;
    int row = blockIdx.x % IMG;
    int bo  = blockIdx.x / IMG;
    int o   = bo % QKV;
    int b   = bo / QKV;
    const __bf16* inp = pre + (size_t)bo * HW;
    const float* w = dw_w + (size_t)o * 9;
    float acc = 0.f;
#pragma unroll
    for (int dy = -1; dy <= 1; dy++) {
        int r = row + dy;
        if ((unsigned)r < IMG) {
            const __bf16* rp = inp + (size_t)r * IMG;
#pragma unroll
            for (int dx = -1; dx <= 1; dx++) {
                int cc = col + dx;
                if ((unsigned)cc < IMG)
                    acc += w[(dy + 1) * 3 + (dx + 1)] * (float)rp[cc];
            }
        }
    }
    out[(size_t)bo * HW + (size_t)row * IMG + col] = (__bf16)acc;
    if (o < 2 * DIM) {
        float sq = acc * acc;
#pragma unroll
        for (int off = 16; off > 0; off >>= 1) sq += __shfl_xor(sq, off, 32);
        int lane = threadIdx.x & 31, wv = threadIdx.x >> 5;
        if (lane == 0) red[wv] = sq;
        __syncthreads();
        if (threadIdx.x == 0)
            atomicAdd(&norms[(size_t)b * 2 * DIM + o], red[0] + red[1] + red[2] + red[3]);
    }
}

// =============== Kernel 3: S = Q K^T over n=16384 (WMMA), normalize, softmax ===============
__global__ void attn_softmax_kernel(const __bf16* __restrict__ qkv, const float* __restrict__ norms,
                                    const float* __restrict__ temperature, float* __restrict__ attn) {
    __shared__ float lds[8 * 1024];
    __shared__ float S[CH * CH];
    __shared__ float P[CH * CH];
    int bh = blockIdx.x, h = bh % HEADS, b = bh / HEADS;
    int lane = threadIdx.x & 31, wave = threadIdx.x >> 5;
    const __bf16* qb = qkv + ((size_t)b * QKV + h * CH) * HW;
    const __bf16* kb = qkv + ((size_t)b * QKV + DIM + h * CH) * HW;
    v8f z = {0.f,0.f,0.f,0.f,0.f,0.f,0.f,0.f};
    v8f acc[2][2] = {{z, z}, {z, z}};
    for (int nb = wave * 32; nb < HW; nb += 8 * 32) {
        v16bf a0 = load_a_bf16_m(qb + nb, HW, lane, 16);
        v16bf a1 = load_a_bf16_m(qb + (size_t)16 * HW + nb, HW, lane, 8);
        v16bf b0 = load_bT_bf16_m(kb + nb, HW, lane, 16);
        v16bf b1 = load_bT_bf16_m(kb + (size_t)16 * HW + nb, HW, lane, 8);
        acc[0][0] = wmma_bf16(a0, b0, acc[0][0]);
        acc[0][1] = wmma_bf16(a0, b1, acc[0][1]);
        acc[1][0] = wmma_bf16(a1, b0, acc[1][0]);
        acc[1][1] = wmma_bf16(a1, b1, acc[1][1]);
    }
#pragma unroll
    for (int ti = 0; ti < 2; ti++)
#pragma unroll
        for (int tj = 0; tj < 2; tj++)
#pragma unroll
            for (int j = 0; j < 8; j++)
                lds[wave * 1024 + (ti * 2 + tj) * 256 + j * 32 + lane] = acc[ti][tj][j];
    __syncthreads();
    for (int s = threadIdx.x; s < 1024; s += 256) {
        float sum = 0.f;
#pragma unroll
        for (int w2 = 0; w2 < 8; w2++) sum += lds[w2 * 1024 + s];
        int tile = s >> 8, j = (s >> 5) & 7, ln = s & 31;
        int c = (tile >> 1) * 16 + j + 8 * (ln >> 4);
        int d = (tile & 1) * 16 + (ln & 15);
        if (c < CH && d < CH) S[c * CH + d] = sum;
    }
    __syncthreads();
    float temp = temperature[h];
    for (int t = threadIdx.x; t < CH * CH; t += 256) {
        int c = t / CH, d = t % CH;
        float qn = fmaxf(sqrtf(norms[(size_t)b * 2 * DIM + h * CH + c]), 1e-12f);
        float kn = fmaxf(sqrtf(norms[(size_t)b * 2 * DIM + DIM + h * CH + d]), 1e-12f);
        P[t] = S[t] * temp / (qn * kn);
    }
    __syncthreads();
    if (threadIdx.x < CH) {
        int c = threadIdx.x;
        float mx = -1e30f;
        for (int d = 0; d < CH; d++) mx = fmaxf(mx, P[c * CH + d]);
        float e[CH], ssum = 0.f;
        for (int d = 0; d < CH; d++) { e[d] = __expf(P[c * CH + d] - mx); ssum += e[d]; }
        float inv = 1.f / ssum;
        for (int d = 0; d < CH; d++)
            attn[((size_t)bh * CH + c) * CH + d] = e[d] * inv;
    }
}

// =============== Kernel 4: fold softmax into projection ===============
__global__ void fold_proj_kernel(const float* __restrict__ proj_w, const float* __restrict__ attn,
                                 __bf16* __restrict__ Wp) {
    __shared__ float A[CH * CH];
    int bh = blockIdx.x, h = bh % HEADS, b = bh / HEADS;
    for (int t = threadIdx.x; t < CH * CH; t += 256) A[t] = attn[(size_t)bh * CH * CH + t];
    __syncthreads();
    for (int t = threadIdx.x; t < DIM * CH; t += 256) {
        int o = t / CH, d = t % CH;
        float s = 0.f;
#pragma unroll
        for (int c = 0; c < CH; c++) s += proj_w[(size_t)o * DIM + h * CH + c] * A[c * CH + d];
        Wp[((size_t)b * DIM + o) * DIM + h * CH + d] = (__bf16)s;
    }
}

// =============== Kernel 5: out = W'_b (192x192) @ V_b (192x16384), fp32 out ===============
// grid: (HW/64, 3, BATCH), block 128 (4 waves). B panel (192x64 bf16, 24 KB) staged by TDM.
__global__ void out_gemm_kernel(const __bf16* __restrict__ Wp, const __bf16* __restrict__ qkv,
                                float* __restrict__ out) {
    __shared__ __bf16 Bpanel[DIM * 64];         // 24 KB
    int lane = threadIdx.x & 31, wave = threadIdx.x >> 5;
    int mtile = blockIdx.y * 4 + wave;          // 0..11
    int o0 = mtile * 16;
    int n0 = blockIdx.x * 64;
    int b  = blockIdx.z;
    const __bf16* Ab = Wp + (size_t)b * DIM * DIM + (size_t)o0 * DIM;
    const __bf16* gsrc = qkv + ((size_t)b * QKV + 2 * DIM) * HW + n0;   // V channels
    if (threadIdx.x == 0) {
        tdm_load_2d(lds_off_u32(Bpanel), gsrc, (unsigned)HW, (unsigned)DIM,
                    64u, (unsigned)DIM, (unsigned long long)HW, 1);
        __builtin_amdgcn_s_wait_tensorcnt((short)0);
    }
    __builtin_prefetch(Ab, 0, 3);
    __syncthreads();
    v8f z = {0.f,0.f,0.f,0.f,0.f,0.f,0.f,0.f};
    v8f acc[4] = {z, z, z, z};
    for (int k0 = 0; k0 < DIM; k0 += 32) {
        v16bf a = load_a_bf16(Ab + k0, DIM, lane);
#pragma unroll
        for (int t = 0; t < 4; t++) {
            v16bf bb = load_b_lds_bf16(&Bpanel[k0 * 64 + t * 16], 64, lane);
            acc[t] = wmma_bf16(a, bb, acc[t]);
        }
    }
    int n = lane & 15, mb = (lane >> 4) * 8;
    float* op = out + ((size_t)b * DIM + o0) * HW + n0;
#pragma unroll
    for (int t = 0; t < 4; t++)
#pragma unroll
        for (int j = 0; j < 8; j++)
            op[(size_t)(mb + j) * HW + t * 16 + n] = acc[t][j];
}

extern "C" void kernel_launch(void* const* d_in, const int* in_sizes, int n_in,
                              void* d_out, int out_size, void* d_ws, size_t ws_size,
                              hipStream_t stream) {
    const float* x           = (const float*)d_in[0];
    const float* y           = (const float*)d_in[1];
    const float* qkv_w       = (const float*)d_in[2];
    const float* dw_w        = (const float*)d_in[3];
    const float* proj_w      = (const float*)d_in[4];
    const float* temperature = (const float*)d_in[5];
    float* out = (float*)d_out;

    char* ws = (char*)d_ws;
    size_t szPre = (size_t)BATCH * QKV * HW * sizeof(__bf16);     // 75.5 MB
    __bf16* pre   = (__bf16*)ws;  ws += szPre;
    __bf16* qkvd  = (__bf16*)ws;  ws += szPre;
    float*  norms = (float*)ws;   ws += (size_t)BATCH * 2 * DIM * sizeof(float);
    float*  attn  = (float*)ws;   ws += (size_t)BATCH * HEADS * CH * CH * sizeof(float);
    __bf16* Wp    = (__bf16*)ws;  ws += (size_t)BATCH * DIM * DIM * sizeof(__bf16);

    hipMemsetAsync(norms, 0, (size_t)BATCH * 2 * DIM * sizeof(float), stream);

    dim3 g1(HW / 64, 9, BATCH);
    qkv_gemm_kernel<<<g1, 128, 0, stream>>>(x, y, qkv_w, pre);

    dwconv_kernel<<<BATCH * QKV * IMG, 128, 0, stream>>>(pre, dw_w, qkvd, norms);

    attn_softmax_kernel<<<BATCH * HEADS, 256, 0, stream>>>(qkvd, norms, temperature, attn);

    fold_proj_kernel<<<BATCH * HEADS, 256, 0, stream>>>(proj_w, attn, Wp);

    dim3 g5(HW / 64, 3, BATCH);
    out_gemm_kernel<<<g5, 128, 0, stream>>>(Wp, qkvd, out);
}
